// CapsGATattentionGRU_28552942584476
// MI455X (gfx1250) — matrix operand order repaired and checked by reading, not compiled, not run on hardware
//
#include <hip/hip_runtime.h>
#include <hip/hip_bf16.h>

// ---------------------------------------------------------------------------
// Problem constants (from reference): D=32, T=32, H=64, B=128, F=2048, G=6144
// ---------------------------------------------------------------------------
#define FDIM 2048
#define GDIM 6144            // 3*F
#define TDIM 32
#define BDIM 128
#define HDIM 64
#define NNODE 4096           // B*D
#define STEP_ELEMS (32*FDIM) // 65536, one scan-step h / ys slab
#define APAD 68              // padded LDS row stride (floats): 68 % 64 banks = 4

typedef __bf16 bf16_t;
typedef __attribute__((ext_vector_type(16))) __bf16        v16bf;
typedef __attribute__((ext_vector_type(8)))  float         v8f;
typedef __attribute__((ext_vector_type(4)))  unsigned int  u32x4;
typedef __attribute__((ext_vector_type(8)))  int           i32x8;
typedef __attribute__((ext_vector_type(4)))  int           i32x4;

#if __has_builtin(__builtin_amdgcn_tensor_load_to_lds) && \
    __has_builtin(__builtin_amdgcn_s_wait_tensorcnt)
#define USE_TDM 1
#else
#define USE_TDM 0
#endif

// ---------------------------------------------------------------------------
// WMMA fragment loaders (bf16 in-register convert from fp32).
// 16-bit A 16x32 layout (ISA 7.12.2): lane L: row=L%16, half=L/16;
//   elems 0..7  -> K = k0 + 8*half + j ; elems 8..15 -> K = k0 + 16 + 8*half + j
// B = W^T, W (N x K) row-major: lane L: col=L%16, same K mapping as A
// ---------------------------------------------------------------------------
__device__ __forceinline__ v16bf load_frag_A(const float* __restrict__ X,
                                             int ld, int m0, int k0, int lane) {
  const int row  = m0 + (lane & 15);
  const int half = lane >> 4;
  const float* p0 = X + (size_t)row * ld + k0 + 8 * half;
  const float* p1 = p0 + 16;
  v16bf a;
#pragma unroll
  for (int j = 0; j < 8; ++j) a[j] = (bf16_t)p0[j];
#pragma unroll
  for (int j = 0; j < 8; ++j) a[8 + j] = (bf16_t)p1[j];
  return a;
}

// Same layout but from a padded LDS panel (row stride APAD floats).
__device__ __forceinline__ v16bf load_frag_A_lds(const float* As, int m0, int kk, int lane) {
  const int row  = m0 + (lane & 15);
  const int half = lane >> 4;
  const float* p0 = As + row * APAD + kk + 8 * half;
  const float* p1 = p0 + 16;
  v16bf a;
#pragma unroll
  for (int j = 0; j < 8; ++j) a[j] = (bf16_t)p0[j];
#pragma unroll
  for (int j = 0; j < 8; ++j) a[8 + j] = (bf16_t)p1[j];
  return a;
}

__device__ __forceinline__ v16bf load_frag_B(const float* __restrict__ W,
                                             int ld, int n0, int k0, int lane) {
  const int col  = n0 + (lane & 15);
  const int half = lane >> 4;
  const float* p = W + (size_t)col * ld + k0 + 8 * half;
  v16bf b;
#pragma unroll
  for (int j = 0; j < 16; ++j) b[j] = (bf16_t)p[(j < 8) ? j : (j + 8)];
  return b;
}

// Generic: one wave -> one 16x16 tile of Y = X @ W^T + bias (used for small GEMMs)
__device__ __forceinline__ void gemm_tile(const float* __restrict__ X,
                                          const float* __restrict__ W,
                                          const float* __restrict__ bias,
                                          float* __restrict__ Y,
                                          int M, int K, int Nout,
                                          int wave, int lane) {
  const int ntiles = Nout >> 4;
  const int mt = wave / ntiles;
  const int nt = wave - mt * ntiles;
  if (mt >= (M >> 4)) return;                  // wave-uniform exit
  const int m0 = mt << 4, n0 = nt << 4;
  v8f c = {};
  for (int k0 = 0; k0 < K; k0 += 32) {
    v16bf a = load_frag_A(X, K, m0, k0, lane);
    v16bf b = load_frag_B(W, K, n0, k0, lane);
    c = __builtin_amdgcn_wmma_f32_16x16x32_bf16(false, a, false, b,
                                                (short)0, c, false, false);
  }
  const int col   = n0 + (lane & 15);
  const float bv  = bias ? bias[col] : 0.0f;
  const int rbase = m0 + ((lane >> 4) << 3);
#pragma unroll
  for (int v = 0; v < 8; ++v)
    Y[(size_t)(rbase + v) * Nout + col] = c[v] + bv;
}

__global__ void k_gemm_bias(const float* __restrict__ X, const float* __restrict__ W,
                            const float* __restrict__ bias, float* __restrict__ Y,
                            int M, int K, int Nout) {
  const int wave = blockIdx.x * (blockDim.x >> 5) + (threadIdx.x >> 5);
  gemm_tile(X, W, bias, Y, M, K, Nout, wave, threadIdx.x & 31);
}

// ---------------------------------------------------------------------------
// GRU step GEMM: Y(32 x 6144) = Xs(32 x 2048) @ W^T + bias.
// Block = 256 threads (8 waves) owns a 32 x 128 output panel.
// A panel (32 x 64 fp32, bank-pad to 68) staged in LDS via TDM
// (tensor_load_to_lds, D# per ISA ch.8, pad_interval=64dw pad_amount=4dw),
// fallback: cooperative global->LDS copy. Each wave: 1 A frag -> 2 WMMAs.
// blockIdx.y: 0 -> gi = xt@Wih^T + bih ; 1 -> gh = h@Whh^T + bhh
// ---------------------------------------------------------------------------
__global__ void k_gru_step_gemm(const float* __restrict__ xt, const float* __restrict__ h,
                                const float* __restrict__ Wih, const float* __restrict__ Whh,
                                const float* __restrict__ bih, const float* __restrict__ bhh,
                                float* __restrict__ gi, float* __restrict__ gh) {
  __shared__ float As[32 * APAD];

  const float* Xs   = (blockIdx.y == 0) ? xt  : h;
  const float* Wsel = (blockIdx.y == 0) ? Wih : Whh;
  const float* bsel = (blockIdx.y == 0) ? bih : bhh;
  float*       Ysel = (blockIdx.y == 0) ? gi  : gh;

  const int lane = threadIdx.x & 31;
  const int w    = threadIdx.x >> 5;       // 0..7
  const int m0   = (w & 1) << 4;           // 0 or 16
  const int n0   = blockIdx.x * 128 + (w >> 1) * 32;  // two adjacent 16-col tiles

  v8f c0 = {};
  v8f c1 = {};

  for (int k0 = 0; k0 < FDIM; k0 += 64) {
#if USE_TDM
    if (threadIdx.x == 0) {
      // D# group0: count=1, lds_addr, global_addr(57b), type=2
      const unsigned ldsOff = (unsigned)(uintptr_t)(void*)As;
      const unsigned long long ga = (unsigned long long)(uintptr_t)(Xs + k0);
      u32x4 g0 = { 1u, ldsOff, (unsigned)ga,
                   (unsigned)((ga >> 32) & 0x01FFFFFFull) | 0x80000000u };
      // D# group1: data_size=4B, pad_enable, pad_interval=64dw, pad_amount=4dw,
      // tensor_dim0=2048, tensor_dim1=32, tile_dim0=64, tile_dim1=32, stride0=2048
      i32x8 g1 = { 0x07520000, 0x08000000, (32 << 16), (64 << 16),
                   32, 2048, 0, 0 };
      i32x4 z4 = { 0, 0, 0, 0 };
      i32x8 z8 = { 0, 0, 0, 0, 0, 0, 0, 0 };
      __builtin_amdgcn_tensor_load_to_lds(g0, g1, z4, z4, z8, 0);
      __builtin_amdgcn_s_wait_tensorcnt(0);
    }
#else
#pragma unroll
    for (int e = threadIdx.x; e < 32 * 64; e += 256)
      As[(e >> 6) * APAD + (e & 63)] = Xs[(size_t)(e >> 6) * FDIM + k0 + (e & 63)];
#endif
    __syncthreads();

#pragma unroll
    for (int kk = 0; kk < 64; kk += 32) {
      v16bf a  = load_frag_A_lds(As, m0, kk, lane);
      v16bf b0 = load_frag_B(Wsel, FDIM, n0,      k0 + kk, lane);
      v16bf b1 = load_frag_B(Wsel, FDIM, n0 + 16, k0 + kk, lane);
      if (kk == 0 && k0 + 64 < FDIM)   // pull next B K-slice toward L2/L0
        __builtin_prefetch(Wsel + (size_t)(n0 + (lane & 15)) * FDIM + k0 + 64, 0, 1);
      c0 = __builtin_amdgcn_wmma_f32_16x16x32_bf16(false, a, false, b0,
                                                   (short)0, c0, false, false);
      c1 = __builtin_amdgcn_wmma_f32_16x16x32_bf16(false, a, false, b1,
                                                   (short)0, c1, false, false);
    }
    __syncthreads();
  }

  const int rbase = m0 + ((lane >> 4) << 3);
  const int colA  = n0 + (lane & 15);
  const int colB  = colA + 16;
  const float bvA = bsel[colA], bvB = bsel[colB];
#pragma unroll
  for (int v = 0; v < 8; ++v) {
    Ysel[(size_t)(rbase + v) * GDIM + colA] = c0[v] + bvA;
    Ysel[(size_t)(rbase + v) * GDIM + colB] = c1[v] + bvB;
  }
}

// ---------------------------------------------------------------------------
// Elementwise / reduction kernels
// ---------------------------------------------------------------------------
__global__ void k_gru_gates(const float* __restrict__ gi, const float* __restrict__ gh,
                            float* __restrict__ h, float* __restrict__ ys) {
  const int idx = blockIdx.x * blockDim.x + threadIdx.x;
  if (idx >= STEP_ELEMS) return;
  const int t = idx >> 11;
  const int f = idx & (FDIM - 1);
  const size_t rb = (size_t)t * GDIM;
  const float ir = gi[rb + f],          hr = gh[rb + f];
  const float iz = gi[rb + FDIM + f],   hz = gh[rb + FDIM + f];
  const float in = gi[rb + 2*FDIM + f], hn = gh[rb + 2*FDIM + f];
  const float r = 1.0f / (1.0f + __expf(-(ir + hr)));
  const float z = 1.0f / (1.0f + __expf(-(iz + hz)));
  const float n = tanhf(in + r * hn);
  const float hnew = (1.0f - z) * n + z * h[idx];
  h[idx]  = hnew;
  ys[idx] = hnew;
}

__global__ void k_zero(float* __restrict__ p, int n) {
  const int i = blockIdx.x * blockDim.x + threadIdx.x;
  if (i < n) p[i] = 0.0f;
}

__global__ void k_nan_clean(const float* __restrict__ in, float* __restrict__ out, int n) {
  const int i = blockIdx.x * blockDim.x + threadIdx.x;
  if (i >= n) return;
  float v = in[i];
  const float FMAX = 3.4028234663852886e38f;
  if (v != v) v = 0.0f;
  else if (v >  FMAX) v = 1.0f;      // posinf -> 1.0 (per reference)
  else if (v < -FMAX) v = -FMAX;     // neginf -> -float_max
  out[i] = v;
}

__global__ void k_temporal_attn(const float* __restrict__ emb,
                                const float* __restrict__ A_w,
                                const float* __restrict__ A_b,
                                float* __restrict__ xn) {
  __shared__ float sAw[TDIM * TDIM];
  __shared__ float sAb[TDIM];
  for (int i = threadIdx.x; i < TDIM * TDIM; i += blockDim.x) sAw[i] = A_w[i];
  if (threadIdx.x < TDIM) sAb[threadIdx.x] = A_b[threadIdx.x];
  __syncthreads();
  const int idx = blockIdx.x * blockDim.x + threadIdx.x;
  if (idx >= BDIM * FDIM) return;
  const int b = idx >> 11, f = idx & (FDIM - 1);
  float a[TDIM];
#pragma unroll
  for (int t = 0; t < TDIM; ++t)
    a[t] = tanhf(emb[(size_t)(b * TDIM + t) * FDIM + f]);
  float aw[TDIM], m = -3.4e38f;
#pragma unroll
  for (int u = 0; u < TDIM; ++u) {
    float s = sAb[u];
#pragma unroll
    for (int t = 0; t < TDIM; ++t) s += a[t] * sAw[u * TDIM + t];
    aw[u] = s; m = fmaxf(m, s);
  }
  float den = 0.0f, vec = 0.0f;
#pragma unroll
  for (int u = 0; u < TDIM; ++u) {
    const float e = __expf(aw[u] - m);
    den += e; vec += e * a[u];
  }
  xn[idx] = tanhf(vec / den);
}

// GATv2 over dense 32-node graphs. Wave per destination node (b,j); lane = source i.
__global__ void k_gat_attn(const float* __restrict__ xl, const float* __restrict__ xr,
                           const float* __restrict__ att, const float* __restrict__ bias,
                           float* __restrict__ out) {
  const int wid = blockIdx.x * (blockDim.x >> 5) + (threadIdx.x >> 5);
  if (wid >= NNODE) return;
  const int lane = threadIdx.x & 31;
  const int b = wid >> 5, j = wid & 31;
  const float* xli = xl + (size_t)(b * 32 + lane) * HDIM;
  const float* xrj = xr + (size_t)(b * 32 + j) * HDIM;
  float logit = 0.0f;
#pragma unroll 8
  for (int hh = 0; hh < HDIM; ++hh) {
    float e = xli[hh] + xrj[hh];
    e = (e > 0.0f) ? e : 0.2f * e;          // leaky_relu(0.2)
    logit += e * att[hh];
  }
  float m = logit;
  for (int o = 16; o; o >>= 1) m = fmaxf(m, __shfl_xor(m, o, 32));
  const float ex = __expf(logit - m);
  float den = ex;
  for (int o = 16; o; o >>= 1) den += __shfl_xor(den, o, 32);
  const float alpha = ex / den;
  float acc0 = 0.0f, acc1 = 0.0f;
  for (int i = 0; i < 32; ++i) {
    const float ai = __shfl(alpha, i, 32);
    const float* xs = xl + (size_t)(b * 32 + i) * HDIM;
    acc0 += ai * xs[lane];
    acc1 += ai * xs[lane + 32];
  }
  const size_t row = (size_t)(b * 32 + j) * HDIM;
  out[row + lane]      = tanhf(acc0 + bias[lane]);
  out[row + lane + 32] = tanhf(acc1 + bias[lane + 32]);
}

__global__ void k_fusion(const float* __restrict__ xn, const float* __restrict__ g0,
                         const float* __restrict__ g1, float* __restrict__ fus) {
  const int idx = blockIdx.x * blockDim.x + threadIdx.x;
  if (idx >= NNODE * 2 * HDIM) return;
  const int row = idx >> 7, k = idx & 127;
  const size_t r64 = (size_t)row * HDIM;
  fus[idx] = (k < HDIM) ? xn[r64 + k] : (g0[r64 + k - HDIM] + g1[r64 + k - HDIM]);
}

// Dynamic routing (3 iters). Wave per (n,l); lane = capsule input c.
__global__ void k_routing(const float* __restrict__ priors, float* __restrict__ caps) {
  const int wid = blockIdx.x * (blockDim.x >> 5) + (threadIdx.x >> 5);
  if (wid >= BDIM * HDIM) return;
  const int lane = threadIdx.x & 31;          // c
  const int n = wid >> 6, l = wid & 63;
  float pr[32], lg[32], outv[32];
#pragma unroll
  for (int o = 0; o < 32; ++o) {
    pr[o] = priors[(size_t)(n * 32 + lane) * FDIM + o * HDIM + l];
    lg[o] = 0.0f;
  }
  for (int r = 0; r < 3; ++r) {
    float m = lg[0];
#pragma unroll
    for (int o = 1; o < 32; ++o) m = fmaxf(m, lg[o]);
    float p[32], den = 0.0f;
#pragma unroll
    for (int o = 0; o < 32; ++o) { p[o] = __expf(lg[o] - m); den += p[o]; }
    const float inv = 1.0f / den;
#pragma unroll
    for (int o = 0; o < 32; ++o) {
      float v = p[o] * inv * pr[o];
      for (int off = 16; off; off >>= 1) v += __shfl_xor(v, off, 32);
      outv[o] = v;
      if (r < 2) lg[o] += pr[o] * v;
    }
  }
  if (lane == 0) {
#pragma unroll
    for (int o = 0; o < 32; ++o)
      caps[(size_t)(n * 32 + o) * HDIM + l] = outv[o];
  }
}

__global__ void k_final(const float* __restrict__ caps, const float* __restrict__ F_w,
                        const float* __restrict__ F_b, float* __restrict__ out) {
  const int idx = blockIdx.x * blockDim.x + threadIdx.x;
  if (idx >= NNODE * 32) return;
  const int r = idx >> 5, d = idx & 31;
  float s = F_b[d];
#pragma unroll 8
  for (int l = 0; l < HDIM; ++l)
    s += tanhf(caps[(size_t)r * HDIM + l]) * F_w[d * HDIM + l];
  out[idx] = tanhf(s);
}

// ---------------------------------------------------------------------------
// Host launch
// ---------------------------------------------------------------------------
static inline int ceil_div(int a, int b) { return (a + b - 1) / b; }

static void launch_gemm(const float* X, const float* W, const float* bias, float* Y,
                        int M, int K, int Nout, hipStream_t stream) {
  const int tiles  = (M >> 4) * (Nout >> 4);
  const int blocks = ceil_div(tiles, 8);
  k_gemm_bias<<<blocks, 256, 0, stream>>>(X, W, bias, Y, M, K, Nout);
}

extern "C" void kernel_launch(void* const* d_in, const int* in_sizes, int n_in,
                              void* d_out, int out_size, void* d_ws, size_t ws_size,
                              hipStream_t stream) {
  const float* inputs = (const float*)d_in[0];
  const float* Wih0 = (const float*)d_in[1];
  const float* Whh0 = (const float*)d_in[2];
  const float* bih0 = (const float*)d_in[3];
  const float* bhh0 = (const float*)d_in[4];
  const float* Wih1 = (const float*)d_in[5];
  const float* Whh1 = (const float*)d_in[6];
  const float* bih1 = (const float*)d_in[7];
  const float* bhh1 = (const float*)d_in[8];
  const float* A_w  = (const float*)d_in[9];
  const float* A_b  = (const float*)d_in[10];
  const float* g0_Wl = (const float*)d_in[11];
  const float* g0_bl = (const float*)d_in[12];
  const float* g0_Wr = (const float*)d_in[13];
  const float* g0_br = (const float*)d_in[14];
  const float* g0_att = (const float*)d_in[15];
  const float* g0_bias = (const float*)d_in[16];
  const float* g1_Wl = (const float*)d_in[17];
  const float* g1_bl = (const float*)d_in[18];
  const float* g1_Wr = (const float*)d_in[19];
  const float* g1_br = (const float*)d_in[20];
  const float* g1_att = (const float*)d_in[21];
  const float* g1_bias = (const float*)d_in[22];
  const float* W_caps = (const float*)d_in[23];   // (32,64,128) == (2048,128) row-major
  const float* F_w  = (const float*)d_in[24];
  const float* F_b  = (const float*)d_in[25];
  // d_in[26] = edge_index: statically dense all-pairs per 32-node block.

  float* out = (float*)d_out;

  const size_t SLAB = (size_t)BDIM * TDIM * FDIM;   // 8,388,608 floats
  float* ws   = (float*)d_ws;
  float* Xc   = ws;
  float* ys0  = Xc  + SLAB;
  float* emb  = ys0 + SLAB;
  float* priors = Xc;                // reuse after layer-0 recurrence
  float* sm   = emb + SLAB;
  float* hbuf = sm;                  sm += STEP_ELEMS;
  float* gi   = sm;                  sm += (size_t)32 * GDIM;
  float* gh   = sm;                  sm += (size_t)32 * GDIM;
  float* xn   = sm;                  sm += (size_t)NNODE * HDIM;
  float* xl   = sm;                  sm += (size_t)NNODE * HDIM;
  float* xr   = sm;                  sm += (size_t)NNODE * HDIM;
  float* g0b  = sm;                  sm += (size_t)NNODE * HDIM;
  float* g1b  = sm;                  sm += (size_t)NNODE * HDIM;
  float* fus  = sm;                  sm += (size_t)NNODE * 2 * HDIM;
  float* caps = sm;                  sm += (size_t)NNODE * HDIM;

  // 1) nan_to_num
  k_nan_clean<<<ceil_div((int)SLAB, 256), 256, 0, stream>>>(inputs, Xc, (int)SLAB);

  // 2) GRU layer 0 (scan over axis 0 = B; per-step batch = T = 32 rows)
  const dim3 gru_grid(48, 2);   // 48 blocks x 128 cols = 6144; y selects gi/gh
  k_zero<<<ceil_div(STEP_ELEMS, 256), 256, 0, stream>>>(hbuf, STEP_ELEMS);
  for (int s = 0; s < BDIM; ++s) {
    k_gru_step_gemm<<<gru_grid, 256, 0, stream>>>(Xc + (size_t)s * STEP_ELEMS, hbuf,
                                                  Wih0, Whh0, bih0, bhh0, gi, gh);
    k_gru_gates<<<ceil_div(STEP_ELEMS, 256), 256, 0, stream>>>(gi, gh, hbuf,
                                                  ys0 + (size_t)s * STEP_ELEMS);
  }

  // 3) GRU layer 1
  k_zero<<<ceil_div(STEP_ELEMS, 256), 256, 0, stream>>>(hbuf, STEP_ELEMS);
  for (int s = 0; s < BDIM; ++s) {
    k_gru_step_gemm<<<gru_grid, 256, 0, stream>>>(ys0 + (size_t)s * STEP_ELEMS, hbuf,
                                                  Wih1, Whh1, bih1, bhh1, gi, gh);
    k_gru_gates<<<ceil_div(STEP_ELEMS, 256), 256, 0, stream>>>(gi, gh, hbuf,
                                                  emb + (size_t)s * STEP_ELEMS);
  }

  // 4) temporal attention -> xn (4096 x 64)
  k_temporal_attn<<<ceil_div(BDIM * FDIM, 256), 256, 0, stream>>>(emb, A_w, A_b, xn);

  // 5) GATv2 layer 0
  launch_gemm(xn, g0_Wl, g0_bl, xl, NNODE, HDIM, HDIM, stream);
  launch_gemm(xn, g0_Wr, g0_br, xr, NNODE, HDIM, HDIM, stream);
  k_gat_attn<<<ceil_div(NNODE, 8), 256, 0, stream>>>(xl, xr, g0_att, g0_bias, g0b);

  // 6) GATv2 layer 1
  launch_gemm(g0b, g1_Wl, g1_bl, xl, NNODE, HDIM, HDIM, stream);
  launch_gemm(g0b, g1_Wr, g1_br, xr, NNODE, HDIM, HDIM, stream);
  k_gat_attn<<<ceil_div(NNODE, 8), 256, 0, stream>>>(xl, xr, g1_att, g1_bias, g1b);

  // 7) fusion + capsule priors GEMM (4096 x 2048 = fusion @ W_caps^T)
  k_fusion<<<ceil_div(NNODE * 2 * HDIM, 256), 256, 0, stream>>>(xn, g0b, g1b, fus);
  launch_gemm(fus, W_caps, nullptr, priors, NNODE, 2 * HDIM, FDIM, stream);

  // 8) dynamic routing -> caps
  k_routing<<<ceil_div(BDIM * HDIM, 8), 256, 0, stream>>>(priors, caps);

  // 9) final FC + tanh -> d_out (128 x 32 x 32)
  k_final<<<ceil_div(NNODE * 32, 256), 256, 0, stream>>>(caps, F_w, F_b, out);
  (void)in_sizes; (void)n_in; (void)out_size; (void)ws_size;
}
